// GCNConv_55585466744854
// MI455X (gfx1250) — matrix operand order repaired and checked by reading, compiled, not run on hardware
//
#include <hip/hip_runtime.h>
#include <hip/hip_bf16.h>

// GCNConv on dense adjacency, MI455X (gfx1250, wave32, WMMA).
//
// Sizes: N=8192 nodes, C=512 channels.
// out = LeakyReLU( D^-1/2 (adj + I) D^-1/2 @ [x_u@Wr ; x_v@Wd] + bias )
//
// Pipeline:
//   1) deg_kernel   : dinv[i] = rsqrt(rowsum(adj)+1)              (268 MB HBM read)
//   2) wtrans_kernel: WT[c][k] = bf16(W[k][c])  (both weights)    (tiny)
//   3) feat_kernel  : hsT[c][j] = dinv[j] * (x[j] @ W)[c]  bf16   (WMMA, 4.3 GFLOP)
//   4) spmm_kernel  : out[i][c] = lrelu(dinv[i]*(adj@hsT + hsT[c][i]) + bias)
//                     (WMMA bf16, 68.7 GFLOP, 268 MB adj streamed once;
//                      hsT = 8 MB, L2-resident)

#define NN   8192
#define NU   4096
#define CIN  512
#define COUT 512

typedef __attribute__((ext_vector_type(16))) __bf16 v16bf;
typedef __attribute__((ext_vector_type(8)))  __bf16 v8bf;
typedef __attribute__((ext_vector_type(8)))  float  v8f;

__device__ __forceinline__ v16bf cvt16(float4 f0, float4 f1, float4 f2, float4 f3) {
  v16bf r;
  r[0]  = (__bf16)f0.x; r[1]  = (__bf16)f0.y; r[2]  = (__bf16)f0.z; r[3]  = (__bf16)f0.w;
  r[4]  = (__bf16)f1.x; r[5]  = (__bf16)f1.y; r[6]  = (__bf16)f1.z; r[7]  = (__bf16)f1.w;
  r[8]  = (__bf16)f2.x; r[9]  = (__bf16)f2.y; r[10] = (__bf16)f2.z; r[11] = (__bf16)f2.w;
  r[12] = (__bf16)f3.x; r[13] = (__bf16)f3.y; r[14] = (__bf16)f3.z; r[15] = (__bf16)f3.w;
  return r;
}

__device__ __forceinline__ v16bf cat8(v8bf lo, v8bf hi) {
  v16bf r;
#pragma unroll
  for (int i = 0; i < 8; ++i) { r[i] = lo[i]; r[i + 8] = hi[i]; }
  return r;
}

// ---------------------------------------------------------------------------
// 1) Degree / dinv: one block per row, float4 coalesced.
// ---------------------------------------------------------------------------
__global__ void deg_kernel(const float* __restrict__ adj, float* __restrict__ dinv) {
  __shared__ float red[256];
  const int i = blockIdx.x;
  const float4* row = reinterpret_cast<const float4*>(adj + (size_t)i * NN);
  float s = 0.f;
#pragma unroll
  for (int it = 0; it < 8; ++it) {
    float4 f = row[threadIdx.x + 256 * it];
    s += f.x + f.y + f.z + f.w;
  }
  red[threadIdx.x] = s;
  __syncthreads();
  for (int off = 128; off > 0; off >>= 1) {
    if ((int)threadIdx.x < off) red[threadIdx.x] += red[threadIdx.x + off];
    __syncthreads();
  }
  if (threadIdx.x == 0) dinv[i] = rsqrtf(red[0] + 1.0f);  // self-loop => deg >= 1
}

// ---------------------------------------------------------------------------
// 2) Weight transpose + f32->bf16: WT[c][k] = W[k][c].
// ---------------------------------------------------------------------------
__global__ void wtrans_kernel(const float* __restrict__ wr, const float* __restrict__ wd,
                              __bf16* __restrict__ wrT, __bf16* __restrict__ wdT) {
  int gid = blockIdx.x * 256 + threadIdx.x;  // 0 .. 512*512-1
  int c = gid >> 9, k = gid & 511;
  wrT[c * 512 + k] = (__bf16)wr[k * 512 + c];
  wdT[c * 512 + k] = (__bf16)wd[k * 512 + c];
}

// ---------------------------------------------------------------------------
// 3) Feature transform: hsT[c][j] = dinv[j] * sum_k W[k][c] * x[j][k].
//    WMMA: M = channel c, K = feature k, N = node j.
//    Block = 16 nodes x 512 channels; 8 waves x 4 c-tiles.
// ---------------------------------------------------------------------------
__global__ void feat_kernel(const float* __restrict__ x,
                            const __bf16* __restrict__ wrT,
                            const __bf16* __restrict__ wdT,
                            const float* __restrict__ dinv,
                            __bf16* __restrict__ hsT) {
  const int lane = threadIdx.x & 31;
  const int wave = threadIdx.x >> 5;
  const int lrow = lane & 15;
  const int lhi  = lane >> 4;
  const int jbase = blockIdx.x * 16;
  const int j = jbase + lrow;                       // per-lane node (B column)
  const int cbase = wave * 64;
  const __bf16* WT = (jbase < NU) ? wrT : wdT;      // partition select (uniform)

  v8f acc[4] = {};
  for (int kb = 0; kb < CIN; kb += 32) {
    // B operand: x^T tile (32k x 16j); per lane 16 consecutive k of row j.
    const float* bp = x + (size_t)j * CIN + kb + 16 * lhi;
    v16bf b = cvt16(*(const float4*)(bp),      *(const float4*)(bp + 4),
                    *(const float4*)(bp + 8),  *(const float4*)(bp + 12));
#pragma unroll
    for (int ct = 0; ct < 4; ++ct) {
      // A operand: WT tile (16c x 32k); lane layout K = {lhi*16..+15}.
      const __bf16* ap = WT + (size_t)(cbase + ct * 16 + lrow) * CIN + kb + lhi * 16;
      v16bf a = cat8(*(const v8bf*)(ap), *(const v8bf*)(ap + 8));
      acc[ct] = __builtin_amdgcn_wmma_f32_16x16x32_bf16(
          false, a, false, b, (short)0, acc[ct], false, false);
    }
  }

  const float dj = dinv[j];
#pragma unroll
  for (int ct = 0; ct < 4; ++ct) {
#pragma unroll
    for (int v = 0; v < 8; ++v) {
      int c = cbase + ct * 16 + v + 8 * lhi;        // C/D layout: M = v (+8 for hi half)
      hsT[(size_t)c * NN + j] = (__bf16)(acc[ct][v] * dj);
    }
  }
}

// ---------------------------------------------------------------------------
// 4) Main SpMM-as-GEMM: out' = adj @ hsT^T  (M=i rows, K=j nodes, N=c channels).
//    Block = 32 rows x all 512 channels (adj read from HBM exactly once);
//    8 waves x (2 m-tiles x 4 n-tiles). hsT (8 MB bf16) stays in L2.
// ---------------------------------------------------------------------------
__global__ void spmm_kernel(const float* __restrict__ adj,
                            const __bf16* __restrict__ hsT,
                            const float* __restrict__ dinv,
                            const float* __restrict__ bias,
                            float* __restrict__ out) {
  const int lane = threadIdx.x & 31;
  const int wave = threadIdx.x >> 5;
  const int lrow = lane & 15;
  const int lhi  = lane >> 4;
  const int ibase = blockIdx.x * 32;
  const int cwave = wave * 64;

  v8f acc[2][4] = {};

  const float* aptr0 = adj + (size_t)(ibase + lrow) * NN + lhi * 8;
  const float* aptr1 = adj + (size_t)(ibase + 16 + lrow) * NN + lhi * 8;
  const __bf16* bptr[4];
#pragma unroll
  for (int nt = 0; nt < 4; ++nt)
    bptr[nt] = hsT + (size_t)(cwave + nt * 16 + lrow) * NN + lhi * 8;

  for (int kb = 0; kb < NN; kb += 32) {
    // A operands: adj f32 -> bf16 (exact for the 0/1 adjacency values).
    v16bf a[2];
    {
      const float* p = aptr0 + kb;
      a[0] = cvt16(*(const float4*)(p),      *(const float4*)(p + 4),
                   *(const float4*)(p + 16), *(const float4*)(p + 20));
      p = aptr1 + kb;
      a[1] = cvt16(*(const float4*)(p),      *(const float4*)(p + 4),
                   *(const float4*)(p + 16), *(const float4*)(p + 20));
    }
    if (kb + 512 < NN) {  // prefetch the HBM-streamed operand ahead
      __builtin_prefetch(aptr0 + kb + 512, 0, 1);
      __builtin_prefetch(aptr1 + kb + 512, 0, 1);
    }
#pragma unroll
    for (int nt = 0; nt < 4; ++nt) {
      v16bf b = cat8(*(const v8bf*)(bptr[nt] + kb), *(const v8bf*)(bptr[nt] + kb + 16));  // 2x16B, L2-resident
      acc[0][nt] = __builtin_amdgcn_wmma_f32_16x16x32_bf16(
          false, a[0], false, b, (short)0, acc[0][nt], false, false);
      acc[1][nt] = __builtin_amdgcn_wmma_f32_16x16x32_bf16(
          false, a[1], false, b, (short)0, acc[1][nt], false, false);
    }
  }

  // Epilogue: + self-loop term, * dinv[i], + bias, LeakyReLU(0.01).
#pragma unroll
  for (int mt = 0; mt < 2; ++mt) {
#pragma unroll
    for (int nt = 0; nt < 4; ++nt) {
      const int c = cwave + nt * 16 + lrow;
      const float bc = bias[c];
#pragma unroll
      for (int v = 0; v < 8; ++v) {
        const int i = ibase + mt * 16 + v + 8 * lhi;
        const float selfterm = (float)hsT[(size_t)c * NN + i];
        float val = dinv[i] * (acc[mt][nt][v] + selfterm) + bc;
        val = (val < 0.f) ? 0.01f * val : val;
        out[(size_t)i * COUT + c] = val;
      }
    }
  }
}

// ---------------------------------------------------------------------------
// Launch
// ---------------------------------------------------------------------------
extern "C" void kernel_launch(void* const* d_in, const int* in_sizes, int n_in,
                              void* d_out, int out_size, void* d_ws, size_t ws_size,
                              hipStream_t stream) {
  const float* x    = (const float*)d_in[0];   // [8192, 512]
  const float* adj  = (const float*)d_in[1];   // [8192, 8192]
  const float* wr   = (const float*)d_in[2];   // [512, 512]
  const float* wd   = (const float*)d_in[3];   // [512, 512]
  const float* bias = (const float*)d_in[4];   // [512]
  float* out = (float*)d_out;                  // [8192, 512]

  // Workspace layout (~9.5 MB total):
  char* ws = (char*)d_ws;
  float*  dinv = (float*)ws;                                  // 32 KB
  __bf16* wrT  = (__bf16*)(ws + 32768);                       // 512 KB
  __bf16* wdT  = (__bf16*)(ws + 32768 + 524288);              // 512 KB
  __bf16* hsT  = (__bf16*)(ws + 32768 + 2 * 524288);          // 8 MB  [512][8192]

  deg_kernel  <<<NN, 256, 0, stream>>>(adj, dinv);
  wtrans_kernel<<<(512 * 512) / 256, 256, 0, stream>>>(wr, wd, wrT, wdT);
  feat_kernel <<<NN / 16, 256, 0, stream>>>(x, wrT, wdT, dinv, hsT);
  spmm_kernel <<<NN / 32, 256, 0, stream>>>(adj, hsT, dinv, bias, out);
}
